// BiLSTM_45612552684167
// MI455X (gfx1250) — compile-verified, matrix operand order checked by reading, not run on hardware
//
#include <hip/hip_runtime.h>
#include <hip/hip_bf16.h>
#include <math.h>

typedef __attribute__((ext_vector_type(16))) _Float16 v16h;
typedef __attribute__((ext_vector_type(8)))  _Float16 h8;
typedef __attribute__((ext_vector_type(8)))  float    v8f;

constexpr int HH  = 256;    // lstm hidden
constexpr int G4  = 1024;   // 4*H
constexpr int TT  = 384;    // sequence length (N_R == N_L == 384)
constexpr int DIN = 22;
constexpr int H3  = 1024;

__device__ __forceinline__ float sigmoidf_(float x) { return 1.0f / (1.0f + __expf(-x)); }

// ---------------------------------------------------------------- small utils
__global__ void k_add(const float* __restrict__ a, const float* __restrict__ b,
                      float* __restrict__ o, int n) {
    int i = blockIdx.x * blockDim.x + threadIdx.x;
    if (i < n) o[i] = a[i] + b[i];
}

__global__ void k_cvt_f16(const float* __restrict__ s, _Float16* __restrict__ d, int n) {
    int i = blockIdx.x * blockDim.x + threadIdx.x;
    if (i < n) d[i] = (_Float16)s[i];
}

// W3s[n][k] = W3[n][k] + W3[n][k+512], converted to f16.  (1024 x 512)
__global__ void k_w3s(const float* __restrict__ W3, _Float16* __restrict__ d) {
    int i = blockIdx.x * blockDim.x + threadIdx.x;   // over 1024*512
    int nrow = i >> 9, k = i & 511;
    d[i] = (_Float16)(W3[nrow * 1024 + k] + W3[nrow * 1024 + 512 + k]);
}

// ------------------------------------------------- layer-0 input projection (K=22)
__global__ void k_pre0(const float* __restrict__ vr, const float* __restrict__ vl,
                       const float* __restrict__ Wih0, const float* __restrict__ bsum0,
                       float* __restrict__ pre0) {
    int idx  = blockIdx.x * blockDim.x + threadIdx.x;  // 4 * 384 * 1024
    int n    = idx & 1023;
    int t    = (idx >> 10) % TT;
    int strm = idx / (TT * G4);
    int seq = strm >> 1, dir = strm & 1;
    const float* x = (seq == 0 ? vr : vl) + t * DIN;
    const float* w = Wih0 + (size_t)(dir * G4 + n) * DIN;
    float acc = bsum0[dir * G4 + n];
#pragma unroll
    for (int d = 0; d < DIN; ++d) acc = fmaf(x[d], w[d], acc);
    pre0[idx] = acc;
}

// -------------------------------------------------------------- LSTM recurrence
// One block (1024 threads) per stream.  strm = seq*2 + dir.  dir==1 => reverse.
__global__ __launch_bounds__(1024) void k_lstm_scan(
    const float* __restrict__ pre,   // [4][T][1024]
    const float* __restrict__ Whh,   // [2][1024][256]
    float* __restrict__ out)         // [2][T][512], dir writes cols dir*256..+255
{
    __shared__ float hs[HH];
    __shared__ float gs[G4];
    int strm = blockIdx.x, seq = strm >> 1, dir = strm & 1;
    const float* p = pre + (size_t)strm * TT * G4;
    const float* W = Whh + (size_t)dir * G4 * HH;
    float* o = out + (size_t)seq * TT * (2 * HH) + dir * HH;
    int tid = threadIdx.x;
    float c = 0.0f;
    if (tid < HH) hs[tid] = 0.0f;
    __syncthreads();
    const float4* wrow = (const float4*)(W + (size_t)tid * HH);
    for (int s = 0; s < TT; ++s) {
        int t = dir ? (TT - 1 - s) : s;
        float acc = p[(size_t)t * G4 + tid];
        const float4* h4 = (const float4*)hs;
#pragma unroll 8
        for (int k = 0; k < HH / 4; ++k) {
            float4 w4 = wrow[k];
            float4 hv = h4[k];
            acc = fmaf(w4.x, hv.x, fmaf(w4.y, hv.y, fmaf(w4.z, hv.z, fmaf(w4.w, hv.w, acc))));
        }
        int gate = tid >> 8;                         // 0:i 1:f 2:g 3:o
        gs[tid] = (gate == 2) ? tanhf(acc) : sigmoidf_(acc);
        __syncthreads();
        if (tid < HH) {
            float iv = gs[tid], fv = gs[256 + tid], gg = gs[512 + tid], ov = gs[768 + tid];
            c = fv * c + iv * gg;
            float hn = ov * tanhf(c);
            hs[tid] = hn;
            o[(size_t)t * (2 * HH) + tid] = hn;
        }
        __syncthreads();
    }
}

// ------------------------------------------------------------- WMMA f16 GEMM
// C[M,N] (+bias, relu, *scale) = A[M,K] @ B[N,K]^T ; 8 waves/block, 16x16/wave.
// Fragment gather follows ISA 7.12.2: lane L holds row (L&15); k-subblock by L>>4;
// each fragment = two contiguous 16B (8 x f16) loads.
__global__ __launch_bounds__(256) void k_gemm_wmma(
    const _Float16* __restrict__ A, const _Float16* __restrict__ B,
    const float* __restrict__ bias, float* __restrict__ C,
    int M, int N, int K, int relu, float scale)
{
    int lane = threadIdx.x & 31;
    int wv   = threadIdx.x >> 5;
    int tilesN = N >> 4;
    int tile = blockIdx.x * 8 + wv;
    int m0 = (tile / tilesN) << 4;
    int n0 = (tile % tilesN) << 4;
    int l15 = lane & 15, kh = lane >> 4;
    const _Float16* pa = A + (size_t)(m0 + l15) * K + kh * 8;
    const _Float16* pb = B + (size_t)(n0 + l15) * K + kh * 8;
    v8f c = {};
    for (int k0 = 0; k0 < K; k0 += 32) {
        __builtin_prefetch(pa + k0 + 256, 0, 0);   // global_prefetch_b8
        __builtin_prefetch(pb + k0 + 256, 0, 0);
        h8 al = *(const h8*)(pa + k0);
        h8 ah = *(const h8*)(pa + k0 + 16);
        h8 bl = *(const h8*)(pb + k0);
        h8 bh = *(const h8*)(pb + k0 + 16);
        v16h a, b;
#pragma unroll
        for (int j = 0; j < 8; ++j) { a[j] = al[j]; a[j + 8] = ah[j]; b[j] = bl[j]; b[j + 8] = bh[j]; }
        c = __builtin_amdgcn_wmma_f32_16x16x32_f16(false, a, false, b, (short)0, c, false, false);
    }
    int row0 = m0 + kh * 8;       // C layout: VGPR g -> M = g (+8 for upper half-wave)
    int col  = n0 + l15;
    float bv = bias ? bias[col] : 0.0f;
#pragma unroll
    for (int g = 0; g < 8; ++g) {
        float v = c[g] + bv;
        if (relu) v = fmaxf(v, 0.0f);
        v *= scale;
        C[(size_t)(row0 + g) * N + col] = v;
    }
}

// ------------------------------------------------- fused pairwise interaction
// out[i,j,:] = log_softmax( sum_k relu(u_r[i,k]+u_l[j,k]+b3[k]) * Wout[:,k] + bout )
__global__ __launch_bounds__(128) void k_pairwise(
    const float* __restrict__ u,     // [2][384][1024]; seq0 = r, seq1 = l
    const float* __restrict__ b3, const float* __restrict__ Wout,
    const float* __restrict__ bout, float* __restrict__ out)
{
    __shared__ float sb3[H3], sw0[H3], sw1[H3];
    int tid = threadIdx.x;
    for (int k = tid; k < H3; k += 128) { sb3[k] = b3[k]; sw0[k] = Wout[k]; sw1[k] = Wout[H3 + k]; }
    __syncthreads();
    int i = blockIdx.y;
    int j = blockIdx.x * 128 + tid;
    const float4* ur = (const float4*)(u + (size_t)i * H3);
    const float4* ul = (const float4*)(u + (size_t)TT * H3 + (size_t)j * H3);
    const float4* pb = (const float4*)sb3;
    const float4* p0 = (const float4*)sw0;
    const float4* p1 = (const float4*)sw1;
    float a0 = 0.f, a1 = 0.f;
    for (int k = 0; k < H3 / 4; ++k) {
        float4 r = ur[k], l = ul[k], bb = pb[k], w0 = p0[k], w1 = p1[k];
        float v;
        v = fmaxf(r.x + l.x + bb.x, 0.f); a0 = fmaf(v, w0.x, a0); a1 = fmaf(v, w1.x, a1);
        v = fmaxf(r.y + l.y + bb.y, 0.f); a0 = fmaf(v, w0.y, a0); a1 = fmaf(v, w1.y, a1);
        v = fmaxf(r.z + l.z + bb.z, 0.f); a0 = fmaf(v, w0.z, a0); a1 = fmaf(v, w1.z, a1);
        v = fmaxf(r.w + l.w + bb.w, 0.f); a0 = fmaf(v, w0.w, a0); a1 = fmaf(v, w1.w, a1);
    }
    float l0 = a0 + bout[0], l1 = a1 + bout[1];
    float m = fmaxf(l0, l1);
    float lse = m + logf(__expf(l0 - m) + __expf(l1 - m));
    size_t o = ((size_t)i * TT + j) * 2;
    out[o] = l0 - lse;
    out[o + 1] = l1 - lse;
}

// ---------------------------------------------------------------------- host
extern "C" void kernel_launch(void* const* d_in, const int* in_sizes, int n_in,
                              void* d_out, int out_size, void* d_ws, size_t ws_size,
                              hipStream_t stream) {
    (void)in_sizes; (void)n_in; (void)out_size; (void)ws_size;
    const float* v_r  = (const float*)d_in[0];
    const float* v_l  = (const float*)d_in[1];
    const float* Wih0 = (const float*)d_in[2];
    const float* Whh0 = (const float*)d_in[3];
    const float* bih0 = (const float*)d_in[4];
    const float* bhh0 = (const float*)d_in[5];
    const float* Wih1 = (const float*)d_in[6];
    const float* Whh1 = (const float*)d_in[7];
    const float* bih1 = (const float*)d_in[8];
    const float* bhh1 = (const float*)d_in[9];
    const float* W1   = (const float*)d_in[10];
    const float* b1   = (const float*)d_in[11];
    const float* W2   = (const float*)d_in[12];
    const float* b2   = (const float*)d_in[13];
    const float* W3   = (const float*)d_in[14];
    const float* b3   = (const float*)d_in[15];
    const float* Wout = (const float*)d_in[16];
    const float* bout = (const float*)d_in[17];

    char* ws = (char*)d_ws;
    size_t off = 0;
    auto alloc = [&](size_t bytes) -> char* {
        char* p = ws + off;
        off += (bytes + 255) & ~(size_t)255;
        return p;
    };
    float*    bsum0 = (float*)alloc(2048 * 4);
    float*    bsum1 = (float*)alloc(2048 * 4);
    float*    pre0  = (float*)alloc((size_t)4 * TT * G4 * 4);
    float*    pre1  = (float*)alloc((size_t)4 * TT * G4 * 4);
    float*    x1    = (float*)alloc((size_t)2 * TT * 512 * 4);
    float*    x2    = (float*)alloc((size_t)2 * TT * 512 * 4);
    float*    h1    = (float*)alloc((size_t)2 * TT * 1024 * 4);
    float*    h2    = (float*)alloc((size_t)2 * TT * 512 * 4);
    float*    u     = (float*)alloc((size_t)2 * TT * 1024 * 4);
    _Float16* x1h   = (_Float16*)alloc((size_t)2 * TT * 512 * 2);
    _Float16* x2h   = (_Float16*)alloc((size_t)2 * TT * 512 * 2);
    _Float16* h1h   = (_Float16*)alloc((size_t)2 * TT * 1024 * 2);
    _Float16* h2h   = (_Float16*)alloc((size_t)2 * TT * 512 * 2);
    _Float16* Wih1h = (_Float16*)alloc((size_t)2 * 1024 * 512 * 2);
    _Float16* W1h   = (_Float16*)alloc((size_t)1024 * 512 * 2);
    _Float16* W2h   = (_Float16*)alloc((size_t)512 * 1024 * 2);
    _Float16* W3sh  = (_Float16*)alloc((size_t)1024 * 512 * 2);

    auto g = [](int n) { return (n + 255) / 256; };

    // bias sums + weight conversions
    k_add<<<g(2048), 256, 0, stream>>>(bih0, bhh0, bsum0, 2048);
    k_add<<<g(2048), 256, 0, stream>>>(bih1, bhh1, bsum1, 2048);
    k_cvt_f16<<<g(2 * 1024 * 512), 256, 0, stream>>>(Wih1, Wih1h, 2 * 1024 * 512);
    k_cvt_f16<<<g(1024 * 512), 256, 0, stream>>>(W1, W1h, 1024 * 512);
    k_cvt_f16<<<g(512 * 1024), 256, 0, stream>>>(W2, W2h, 512 * 1024);
    k_w3s<<<g(1024 * 512), 256, 0, stream>>>(W3, W3sh);

    // layer 0: input projection + recurrence
    k_pre0<<<g(4 * TT * G4), 256, 0, stream>>>(v_r, v_l, Wih0, bsum0, pre0);
    k_lstm_scan<<<4, 1024, 0, stream>>>(pre0, Whh0, x1);
    k_cvt_f16<<<g(2 * TT * 512), 256, 0, stream>>>(x1, x1h, 2 * TT * 512);

    // layer 1 input projection: WMMA GEMMs (384x1024x512 each)
    for (int seq = 0; seq < 2; ++seq)
        for (int dir = 0; dir < 2; ++dir)
            k_gemm_wmma<<<192, 256, 0, stream>>>(
                x1h + (size_t)seq * TT * 512, Wih1h + (size_t)dir * 1024 * 512,
                bsum1 + dir * 1024, pre1 + (size_t)(seq * 2 + dir) * TT * 1024,
                384, 1024, 512, 0, 1.0f);
    k_lstm_scan<<<4, 1024, 0, stream>>>(pre1, Whh1, x2);
    k_cvt_f16<<<g(2 * TT * 512), 256, 0, stream>>>(x2, x2h, 2 * TT * 512);

    // MLP1: relu(x2 @ W1^T + b1)   384x1024x512
    for (int seq = 0; seq < 2; ++seq)
        k_gemm_wmma<<<192, 256, 0, stream>>>(
            x2h + (size_t)seq * TT * 512, W1h, b1, h1 + (size_t)seq * TT * 1024,
            384, 1024, 512, 1, 1.0f);
    k_cvt_f16<<<g(2 * TT * 1024), 256, 0, stream>>>(h1, h1h, 2 * TT * 1024);

    // MLP2: relu(h1 @ W2^T + b2)   384x512x1024
    for (int seq = 0; seq < 2; ++seq)
        k_gemm_wmma<<<96, 256, 0, stream>>>(
            h1h + (size_t)seq * TT * 1024, W2h, b2, h2 + (size_t)seq * TT * 512,
            384, 512, 1024, 1, 1.0f);
    k_cvt_f16<<<g(2 * TT * 512), 256, 0, stream>>>(h2, h2h, 2 * TT * 512);

    // u = 0.5 * (h2 @ W3s)         384x1024x512
    for (int seq = 0; seq < 2; ++seq)
        k_gemm_wmma<<<192, 256, 0, stream>>>(
            h2h + (size_t)seq * TT * 512, W3sh, nullptr, u + (size_t)seq * TT * 1024,
            384, 1024, 512, 0, 0.5f);

    // fused pairwise head + log_softmax
    dim3 pg(3, TT);
    k_pairwise<<<pg, 128, 0, stream>>>(u, b3, Wout, bout, (float*)d_out);
}
